// HierarchicalVQVAE_30227979829423
// MI455X (gfx1250) — matrix operand order repaired and checked by reading, not compiled, not run on hardware
//
#include <hip/hip_runtime.h>
#include <hip/hip_bf16.h>
#include <math.h>

// ---------------- CDNA5 WMMA types ----------------
typedef __attribute__((ext_vector_type(16))) __bf16 v16bf;
typedef __attribute__((ext_vector_type(8)))  float  v8f;

#define DMQ   64
#define HIDQ  128
#define FEATQ 96
#define NCQ   256
#define NFQ   1024
#define BETAQ 0.25f
#define WAVES 4

union FragU { v16bf v; uint4 u[2]; };

// A-fragment (16x32 bf16, row-major source, stride in elements).
// ISA layout: lanes 0-15 row M=lane, K = {kb..kb+7, kb+16..kb+23}, kb=0;
// lanes 16-31 same rows, kb=8 interleave.
__device__ __forceinline__ v16bf load_a_frag(const __bf16* base, int stride, int k0, int lane) {
  const int r  = lane & 15;
  const int kb = k0 + ((lane & 16) >> 1);   // 0 or 8
  FragU t;
  t.u[0] = *(const uint4*)(base + r * stride + kb);
  t.u[1] = *(const uint4*)(base + r * stride + kb + 16);
  return t.v;
}

// B-fragment (32x16 bf16) from transposed storage Wt[N][K]:
// lane holds column N=lane&15; K = {kb..kb+7, kb+16..kb+23}, kb=k0 (lanes 0-15) or k0+8 (lanes 16-31).
__device__ __forceinline__ v16bf load_b_frag(const __bf16* base, int stride, int n0, int k0, int lane) {
  const int c  = n0 + (lane & 15);
  const int kb = k0 + ((lane & 16) >> 1);   // 0 or 8
  FragU t;
  t.u[0] = *(const uint4*)(base + c * stride + kb);
  t.u[1] = *(const uint4*)(base + c * stride + kb + 16);
  return t.v;
}

// Preload all A-fragments of a 16xK tile (stride == K) into registers.
template<int K>
__device__ __forceinline__ void load_a_all(const __bf16* A, v16bf* af, int lane) {
  #pragma unroll
  for (int kk = 0; kk < K / 32; ++kk) af[kk] = load_a_frag(A, K, kk * 32, lane);
}

// GEMM 16x16 output tile with A already in registers; B streamed from Wt[N][K].
template<int K>
__device__ __forceinline__ v8f gemm_tile_r(const v16bf* af, const __bf16* Bm, int n0, int lane) {
  v8f acc = {0.f,0.f,0.f,0.f,0.f,0.f,0.f,0.f};
  #pragma unroll
  for (int kk = 0; kk < K / 32; ++kk) {
    v16bf b = load_b_frag(Bm, K, n0, kk * 32, lane);
    acc = __builtin_amdgcn_wmma_f32_16x16x32_bf16(false, af[kk], false, b, (short)0, acc,
                                                  false, false);
  }
  return acc;
}

// Branchless gelu (tanh/sigmoid form): x * sigmoid(1.5957691*(x + 0.044715 x^3)).
// Error ~1e-3, well below the bf16 rounding applied to every activation.
__device__ __forceinline__ float gelu_fast(float x) {
  float x2 = x * x;
  float arg = 1.5957691216057308f * x * __builtin_fmaf(0.044715f, x2, 1.0f);
  // sigmoid(arg) = 1 / (1 + exp2(-arg*log2e))
  float e = __builtin_amdgcn_exp2f(arg * -1.4426950408889634f);
  return x * __builtin_amdgcn_rcpf(1.0f + e);
}

// ---------------- prep kernels ----------------
// dst[n*K + k] = bf16(src[k*N + n])   (weights stored (fan_in=K, fan_out=N))
__global__ void k_transpose_bf16(const float* __restrict__ src, __bf16* __restrict__ dst,
                                 int K, int N) {
  int i = blockIdx.x * blockDim.x + threadIdx.x;
  if (i < K * N) {
    int n = i / K, k = i % K;
    dst[n * K + k] = (__bf16)src[k * N + n];
  }
}

// codebook: already [code][dim] == B-layout; convert + squared norms
__global__ void k_cb_prep(const float* __restrict__ cb, __bf16* __restrict__ cbt,
                          float* __restrict__ e2, int ncodes) {
  int i = blockIdx.x * blockDim.x + threadIdx.x;
  if (i < ncodes) {
    float s = 0.f;
    for (int d = 0; d < DMQ; ++d) {
      float v = cb[i * DMQ + d];
      s += v * v;
      cbt[i * DMQ + d] = (__bf16)v;
    }
    e2[i] = s;
  }
}

// combined head weight [16][128]: rows 0-7 role, 8-9 bounce, 10-11 break, 12-15 zero
__global__ void k_head_prep(const float* __restrict__ Wr, const float* __restrict__ Wbo,
                            const float* __restrict__ Wbr, const float* __restrict__ br,
                            const float* __restrict__ bbo, const float* __restrict__ bbr,
                            __bf16* __restrict__ wh, float* __restrict__ bh) {
  int i = blockIdx.x * blockDim.x + threadIdx.x;
  if (i < 16 * HIDQ) {
    int n = i / HIDQ, k = i % HIDQ;
    float v = (n < 8)  ? Wr[k * 8 + n]
            : (n < 10) ? Wbo[k * 2 + (n - 8)]
            : (n < 12) ? Wbr[k * 2 + (n - 10)] : 0.f;
    wh[n * HIDQ + k] = (__bf16)v;
  }
  if (i < 16) {
    bh[i] = (i < 8) ? br[i] : (i < 10) ? bbo[i - 8] : (i < 12) ? bbr[i - 10] : 0.f;
  }
}

__global__ void k_zero(float* __restrict__ cC, float* __restrict__ cF, float* __restrict__ sums) {
  int t = threadIdx.x;
  if (t < NCQ) cC[t] = 0.f;
  if (t < NFQ) cF[t] = 0.f;
  if (t < 2)   sums[t] = 0.f;
}

// ---------------- fused VQ-VAE kernel (one 16-row tile per wave32) ----------------
__global__ __launch_bounds__(WAVES * 32, 1)
void k_vqvae(const float* __restrict__ feat,
             const __bf16* __restrict__ we1t, const float* __restrict__ be1,
             const __bf16* __restrict__ we2t, const float* __restrict__ be2,
             const __bf16* __restrict__ we3t, const float* __restrict__ be3,
             const float*  __restrict__ cb_c, const __bf16* __restrict__ cbct,
             const float*  __restrict__ e2c,
             const float*  __restrict__ cb_f, const __bf16* __restrict__ cbft,
             const float*  __restrict__ e2f,
             const __bf16* __restrict__ wd1t, const float* __restrict__ bd1,
             const __bf16* __restrict__ wd2t, const float* __restrict__ bd2,
             const __bf16* __restrict__ wft,  const float* __restrict__ bfb,
             const __bf16* __restrict__ wht,  const float* __restrict__ bh,
             float* __restrict__ o_feat, float* __restrict__ o_role,
             float* __restrict__ o_bo,   float* __restrict__ o_brk,
             float* __restrict__ o_z,    float* __restrict__ o_ci, float* __restrict__ o_fi,
             float* __restrict__ countsC, float* __restrict__ countsF,
             float* __restrict__ sums) {
  __shared__ __bf16 smem[WAVES][2][16 * HIDQ];
  const int lane = threadIdx.x & 31;
  const int wid  = threadIdx.x >> 5;
  const int row0 = (blockIdx.x * WAVES + wid) * 16;
  __bf16* bufA = smem[wid][0];
  __bf16* bufB = smem[wid][1];
  const int nl = lane & 15;               // column-within-tile for C/D fragments
  const int mh = (lane & 16) >> 1;        // row half offset (0 or 8)
  v16bf af[4];

  // ---- stage feat tile (16 x 96) -> bufA (bf16) ----
  for (int i = lane; i < 16 * FEATQ; i += 32) {
    int r = i / FEATQ, c = i % FEATQ;
    bufA[r * FEATQ + c] = (__bf16)feat[(size_t)(row0 + r) * FEATQ + c];
  }
  __syncthreads();

  // ---- encoder L1: 16x96 @ (96->128), gelu -> bufB (stride 128) ----
  load_a_all<FEATQ>(bufA, af, lane);
  for (int nt = 0; nt < 8; ++nt) {
    v8f acc = gemm_tile_r<FEATQ>(af, we1t, nt * 16, lane);
    int N = nt * 16 + nl;
    float bv = be1[N];
    #pragma unroll
    for (int j = 0; j < 8; ++j)
      bufB[(j + mh) * HIDQ + N] = (__bf16)gelu_fast(acc[j] + bv);
  }
  __syncthreads();

  // ---- encoder L2: 128->128, gelu -> bufA ----
  load_a_all<HIDQ>(bufB, af, lane);
  for (int nt = 0; nt < 8; ++nt) {
    v8f acc = gemm_tile_r<HIDQ>(af, we2t, nt * 16, lane);
    int N = nt * 16 + nl;
    float bv = be2[N];
    #pragma unroll
    for (int j = 0; j < 8; ++j)
      bufA[(j + mh) * HIDQ + N] = (__bf16)gelu_fast(acc[j] + bv);
  }
  __syncthreads();

  // ---- encoder L3: 128->64 -> z (fp32 regs + bf16 bufB stride 64 + global) ----
  float zr[32];
  load_a_all<HIDQ>(bufA, af, lane);
  #pragma unroll
  for (int nt = 0; nt < 4; ++nt) {
    v8f acc = gemm_tile_r<HIDQ>(af, we3t, nt * 16, lane);
    int N = nt * 16 + nl;
    float bv = be3[N];
    #pragma unroll
    for (int j = 0; j < 8; ++j) {
      float v = acc[j] + bv;
      zr[nt * 8 + j] = v;
      bufB[(j + mh) * DMQ + N] = (__bf16)v;
      o_z[(size_t)(row0 + j + mh) * DMQ + N] = v;
    }
  }
  __syncthreads();

  // ---- coarse VQ: argmin over 256 codes of (e2 - 2 z.e) ----
  float bestd[8]; int besti[8];
  #pragma unroll
  for (int j = 0; j < 8; ++j) { bestd[j] = 3.4e38f; besti[j] = 0; }
  load_a_all<DMQ>(bufB, af, lane);        // z tile, reused for all 16 code tiles
  for (int nt = 0; nt < NCQ / 16; ++nt) {
    v8f acc = gemm_tile_r<DMQ>(af, cbct, nt * 16, lane);
    int N = nt * 16 + nl;
    float e = e2c[N];
    #pragma unroll
    for (int j = 0; j < 8; ++j) {
      float d = e - 2.0f * acc[j];
      if (d < bestd[j]) { bestd[j] = d; besti[j] = N; }
    }
  }
  #pragma unroll
  for (int off = 8; off >= 1; off >>= 1) {
    #pragma unroll
    for (int j = 0; j < 8; ++j) {
      float od = __shfl_xor(bestd[j], off, 32);
      int   oi = __shfl_xor(besti[j], off, 32);
      if (od < bestd[j] || (od == bestd[j] && oi < besti[j])) { bestd[j] = od; besti[j] = oi; }
    }
  }

  // residual (fp32), commit_c partial, residual bf16 -> bufA (stride 64)
  float res[32];
  float csum = 0.f;
  #pragma unroll
  for (int nt = 0; nt < 4; ++nt) {
    int N = nt * 16 + nl;
    #pragma unroll
    for (int j = 0; j < 8; ++j) {
      float q = cb_c[(size_t)besti[j] * DMQ + N];
      float r = zr[nt * 8 + j] - q;
      res[nt * 8 + j] = r;
      csum += r * r;
      bufA[(j + mh) * DMQ + N] = (__bf16)r;
    }
  }
  if (nl == 0) {   // lanes 0 and 16 hold reduced indices for rows 0-7 / 8-15
    #pragma unroll
    for (int j = 0; j < 8; ++j) {
      o_ci[row0 + j + mh] = (float)besti[j];
      atomicAdd(&countsC[besti[j]], 1.0f);
    }
  }
  #pragma unroll
  for (int off = 16; off >= 1; off >>= 1) csum += __shfl_xor(csum, off, 32);
  if (lane == 0) atomicAdd(&sums[0], csum);
  __syncthreads();

  // ---- fine VQ over 1024 codes on residual ----
  #pragma unroll
  for (int j = 0; j < 8; ++j) { bestd[j] = 3.4e38f; besti[j] = 0; }
  load_a_all<DMQ>(bufA, af, lane);        // residual tile, reused for all 64 code tiles
  for (int nt = 0; nt < NFQ / 16; ++nt) {
    if (nt + 1 < NFQ / 16) __builtin_prefetch(cbft + (size_t)(nt + 1) * 16 * DMQ, 0, 1);
    v8f acc = gemm_tile_r<DMQ>(af, cbft, nt * 16, lane);
    int N = nt * 16 + nl;
    float e = e2f[N];
    #pragma unroll
    for (int j = 0; j < 8; ++j) {
      float d = e - 2.0f * acc[j];
      if (d < bestd[j]) { bestd[j] = d; besti[j] = N; }
    }
  }
  #pragma unroll
  for (int off = 8; off >= 1; off >>= 1) {
    #pragma unroll
    for (int j = 0; j < 8; ++j) {
      float od = __shfl_xor(bestd[j], off, 32);
      int   oi = __shfl_xor(besti[j], off, 32);
      if (od < bestd[j] || (od == bestd[j] && oi < besti[j])) { bestd[j] = od; besti[j] = oi; }
    }
  }

  // decoded = cq + fq = (z - res) + fq, commit_f partial; decoded bf16 -> bufB (stride 64)
  float fsum = 0.f;
  #pragma unroll
  for (int nt = 0; nt < 4; ++nt) {
    int N = nt * 16 + nl;
    #pragma unroll
    for (int j = 0; j < 8; ++j) {
      float fq = cb_f[(size_t)besti[j] * DMQ + N];
      float r  = res[nt * 8 + j];
      float df = r - fq;
      fsum += df * df;
      bufB[(j + mh) * DMQ + N] = (__bf16)(zr[nt * 8 + j] - r + fq);
    }
  }
  if (nl == 0) {
    #pragma unroll
    for (int j = 0; j < 8; ++j) {
      o_fi[row0 + j + mh] = (float)besti[j];
      atomicAdd(&countsF[besti[j]], 1.0f);
    }
  }
  #pragma unroll
  for (int off = 16; off >= 1; off >>= 1) fsum += __shfl_xor(fsum, off, 32);
  if (lane == 0) atomicAdd(&sums[1], fsum);
  __syncthreads();

  // ---- decoder L1: 64->128, gelu -> bufA ----
  load_a_all<DMQ>(bufB, af, lane);
  for (int nt = 0; nt < 8; ++nt) {
    v8f acc = gemm_tile_r<DMQ>(af, wd1t, nt * 16, lane);
    int N = nt * 16 + nl;
    float bv = bd1[N];
    #pragma unroll
    for (int j = 0; j < 8; ++j)
      bufA[(j + mh) * HIDQ + N] = (__bf16)gelu_fast(acc[j] + bv);
  }
  __syncthreads();

  // ---- decoder L2: 128->128, gelu -> bufB ----
  load_a_all<HIDQ>(bufA, af, lane);
  for (int nt = 0; nt < 8; ++nt) {
    v8f acc = gemm_tile_r<HIDQ>(af, wd2t, nt * 16, lane);
    int N = nt * 16 + nl;
    float bv = bd2[N];
    #pragma unroll
    for (int j = 0; j < 8; ++j)
      bufB[(j + mh) * HIDQ + N] = (__bf16)gelu_fast(acc[j] + bv);
  }
  __syncthreads();

  // ---- out_feat head (128->96) + fused small heads (128->16) ----
  load_a_all<HIDQ>(bufB, af, lane);
  for (int nt = 0; nt < 6; ++nt) {
    v8f acc = gemm_tile_r<HIDQ>(af, wft, nt * 16, lane);
    int N = nt * 16 + nl;
    float bv = bfb[N];
    #pragma unroll
    for (int j = 0; j < 8; ++j)
      o_feat[(size_t)(row0 + j + mh) * FEATQ + N] = acc[j] + bv;
  }
  {
    v8f acc = gemm_tile_r<HIDQ>(af, wht, 0, lane);
    float bv = bh[nl];
    #pragma unroll
    for (int j = 0; j < 8; ++j) {
      float v = acc[j] + bv;
      int row = row0 + j + mh;
      if (nl < 8)       o_role[(size_t)row * 8 + nl] = v;
      else if (nl < 10) o_bo[(size_t)row * 2 + (nl - 8)] = v;
      else if (nl < 12) o_brk[(size_t)row * 2 + (nl - 10)] = v;
    }
  }
}

// ---------------- finalize scalars ----------------
__global__ void k_finalize(const float* __restrict__ cC, const float* __restrict__ cF,
                           const float* __restrict__ sums, float* __restrict__ o, int Bn) {
  __shared__ float r0[256], r1[256], r2[256], r3[256];
  int t = threadIdx.x;
  float invB = 1.0f / (float)Bn;
  float entc, usedc;
  { float c = cC[t]; float u = c * invB + 1e-10f; entc = -u * logf(u); usedc = (c > 0.f) ? 1.f : 0.f; }
  float entf = 0.f, usedf = 0.f;
  for (int i = t; i < NFQ; i += 256) {
    float c = cF[i]; float u = c * invB + 1e-10f;
    entf += -u * logf(u); usedf += (c > 0.f) ? 1.f : 0.f;
  }
  r0[t] = entc; r1[t] = usedc; r2[t] = entf; r3[t] = usedf;
  __syncthreads();
  for (int s = 128; s > 0; s >>= 1) {
    if (t < s) { r0[t] += r0[t + s]; r1[t] += r1[t + s]; r2[t] += r2[t + s]; r3[t] += r3[t + s]; }
    __syncthreads();
  }
  if (t == 0) {
    float denom = 1.0f / ((float)Bn * (float)DMQ);
    o[0] = BETAQ * sums[0] * denom;   // commit_c
    o[1] = BETAQ * sums[1] * denom;   // commit_f
    o[2] = r0[0];                     // ent_c
    o[3] = r2[0];                     // ent_f
    o[4] = r1[0];                     // used_c
    o[5] = r3[0];                     // used_f
  }
}

// ---------------- host launcher ----------------
extern "C" void kernel_launch(void* const* d_in, const int* in_sizes, int n_in,
                              void* d_out, int out_size, void* d_ws, size_t ws_size,
                              hipStream_t stream) {
  const float* feat = (const float*)d_in[0];
  const float* We1  = (const float*)d_in[1];  const float* be1 = (const float*)d_in[2];
  const float* We2  = (const float*)d_in[3];  const float* be2 = (const float*)d_in[4];
  const float* We3  = (const float*)d_in[5];  const float* be3 = (const float*)d_in[6];
  const float* cb_c = (const float*)d_in[7];  const float* cb_f = (const float*)d_in[8];
  const float* Wd1  = (const float*)d_in[9];  const float* bd1 = (const float*)d_in[10];
  const float* Wd2  = (const float*)d_in[11]; const float* bd2 = (const float*)d_in[12];
  const float* Wf   = (const float*)d_in[13]; const float* bf  = (const float*)d_in[14];
  const float* Wr   = (const float*)d_in[15]; const float* br  = (const float*)d_in[16];
  const float* Wbo  = (const float*)d_in[17]; const float* bbo = (const float*)d_in[18];
  const float* Wbr  = (const float*)d_in[19]; const float* bbr = (const float*)d_in[20];

  const int Bn = in_sizes[0] / FEATQ;

  // workspace carve-up (256B aligned)
  char* ws = (char*)d_ws;
  size_t off = 0;
  auto carve = [&](size_t bytes) { char* p = ws + off; off = (off + bytes + 255) & ~(size_t)255; return p; };
  __bf16* we1t = (__bf16*)carve(HIDQ * FEATQ * 2);
  __bf16* we2t = (__bf16*)carve(HIDQ * HIDQ * 2);
  __bf16* we3t = (__bf16*)carve(DMQ * HIDQ * 2);
  __bf16* wd1t = (__bf16*)carve(HIDQ * DMQ * 2);
  __bf16* wd2t = (__bf16*)carve(HIDQ * HIDQ * 2);
  __bf16* wft  = (__bf16*)carve(FEATQ * HIDQ * 2);
  __bf16* wht  = (__bf16*)carve(16 * HIDQ * 2);
  __bf16* cbct = (__bf16*)carve(NCQ * DMQ * 2);
  __bf16* cbft = (__bf16*)carve(NFQ * DMQ * 2);
  float*  e2c  = (float*)carve(NCQ * 4);
  float*  e2f  = (float*)carve(NFQ * 4);
  float*  bh   = (float*)carve(16 * 4);
  float*  cntC = (float*)carve(NCQ * 4);
  float*  cntF = (float*)carve(NFQ * 4);
  float*  sums = (float*)carve(2 * 4);

  // output sections (flat, in reference return order)
  float* out    = (float*)d_out;
  float* o_feat = out;
  float* o_role = o_feat + (size_t)Bn * FEATQ;
  float* o_bo   = o_role + (size_t)Bn * 8;
  float* o_brk  = o_bo   + (size_t)Bn * 2;
  float* o_z    = o_brk  + (size_t)Bn * 2;
  float* o_ci   = o_z    + (size_t)Bn * DMQ;
  float* o_fi   = o_ci   + (size_t)Bn;
  float* o_scal = o_fi   + (size_t)Bn;

  // --- prep ---
  k_transpose_bf16<<<(FEATQ * HIDQ + 255) / 256, 256, 0, stream>>>(We1, we1t, FEATQ, HIDQ);
  k_transpose_bf16<<<(HIDQ * HIDQ + 255) / 256, 256, 0, stream>>>(We2, we2t, HIDQ, HIDQ);
  k_transpose_bf16<<<(HIDQ * DMQ + 255) / 256, 256, 0, stream>>>(We3, we3t, HIDQ, DMQ);
  k_transpose_bf16<<<(DMQ * HIDQ + 255) / 256, 256, 0, stream>>>(Wd1, wd1t, DMQ, HIDQ);
  k_transpose_bf16<<<(HIDQ * HIDQ + 255) / 256, 256, 0, stream>>>(Wd2, wd2t, HIDQ, HIDQ);
  k_transpose_bf16<<<(HIDQ * FEATQ + 255) / 256, 256, 0, stream>>>(Wf, wft, HIDQ, FEATQ);
  k_head_prep<<<2, 1024, 0, stream>>>(Wr, Wbo, Wbr, br, bbo, bbr, wht, bh);
  k_cb_prep<<<1, 256, 0, stream>>>(cb_c, cbct, e2c, NCQ);
  k_cb_prep<<<4, 256, 0, stream>>>(cb_f, cbft, e2f, NFQ);
  k_zero<<<1, 1024, 0, stream>>>(cntC, cntF, sums);

  // --- fused forward: one 16-row tile per wave, 4 waves per block ---
  k_vqvae<<<Bn / (16 * WAVES), WAVES * 32, 0, stream>>>(
      feat, we1t, be1, we2t, be2, we3t, be3,
      cb_c, cbct, e2c, cb_f, cbft, e2f,
      wd1t, bd1, wd2t, bd2, wft, bf, wht, bh,
      o_feat, o_role, o_bo, o_brk, o_z, o_ci, o_fi,
      cntC, cntF, sums);

  // --- scalars ---
  k_finalize<<<1, 256, 0, stream>>>(cntC, cntF, sums, o_scal, Bn);
}
